// DropoutLSTMLayer_65721589563668
// MI455X (gfx1250) — compile-verified, matrix operand order checked by reading, not run on hardware
//
#include <hip/hip_runtime.h>

// Problem constants (from reference)
#define BB 64
#define SS 512
#define II 1024
#define HH 1024
#define G4 4096   // 4*H

typedef __attribute__((ext_vector_type(16))) _Float16 v16h;
typedef __attribute__((ext_vector_type(8)))  _Float16 v8h;
typedef __attribute__((ext_vector_type(8)))  float    v8f;

union V16U { v16h v; v8h h[2]; };

__device__ __forceinline__ float sigmf(float x) {
    return 1.0f / (1.0f + __expf(-x));
}
__device__ __forceinline__ float tanhfast(float x) {
    x = fminf(15.0f, fmaxf(-15.0f, x));
    float e = __expf(2.0f * x);
    return (e - 1.0f) / (e + 1.0f);
}

// ---------------------------------------------------------------------------
// Convert x [B,S,I] f32 -> X16 [S,B,I] f16  (row r = s*B+b)
// ---------------------------------------------------------------------------
__global__ void conv_x_kernel(const float* __restrict__ x, _Float16* __restrict__ X16) {
    size_t o = (size_t)blockIdx.x * blockDim.x + threadIdx.x;
    if (o >= (size_t)SS * BB * II) return;
    int i = (int)(o & (II - 1));
    size_t r = o >> 10;            // I = 1024
    int b = (int)(r & (BB - 1));   // B = 64
    int s = (int)(r >> 6);
    X16[o] = (_Float16)x[((size_t)b * SS + s) * II + i];
}

// ---------------------------------------------------------------------------
// Pack weight W [4096 x 1024] (row-major, f32) into WMMA-B layout f16:
//   P[nt(256)][kc(32)][lane(32)][16 halves]
//   lane<16 : n = nt*16+lane,    k = kc*32 + {0..7, 16..23}
//   lane>=16: n = nt*16+lane-16, k = kc*32 + {8..15, 24..31}
// ---------------------------------------------------------------------------
__global__ void pack_w_kernel(const float* __restrict__ W, _Float16* __restrict__ P) {
    int idx = blockIdx.x * blockDim.x + threadIdx.x;   // 4096*1024 total
    if (idx >= G4 * HH) return;
    int hh   = idx & 15;
    int lane = (idx >> 4) & 31;
    int kc   = (idx >> 9) & 31;
    int nt   = idx >> 14;
    int n    = nt * 16 + (lane & 15);
    int koff = (hh < 8 ? hh : hh + 8) + ((lane < 16) ? 0 : 8);
    int k    = kc * 32 + koff;
    P[idx] = (_Float16)W[(size_t)n * HH + k];
}

// ---------------------------------------------------------------------------
// Zero recurrent h state (2 dirs x 2 bufs x 64x1024 halves) + grid barrier ctr
// ---------------------------------------------------------------------------
__global__ void init_state_kernel(_Float16* __restrict__ h16, int* __restrict__ bar) {
    int idx = blockIdx.x * blockDim.x + threadIdx.x;
    const int NH = 2 * 2 * BB * HH;
    if (idx < NH) h16[idx] = (_Float16)0.0f;
    if (idx == 0) bar[0] = 0;
}

// ---------------------------------------------------------------------------
// Helpers for the pipelined K-loop body
// ---------------------------------------------------------------------------
#define LOAD_CHUNK(AB, BMB, KC)                                                 \
    do {                                                                        \
        int _k0 = (KC) * 32;                                                    \
        (AB).h[0] = *(const v8h*)(arow + _k0 + aoff);                           \
        (AB).h[1] = *(const v8h*)(arow + _k0 + 16 + aoff);                      \
        _Pragma("unroll")                                                       \
        for (int _q = 0; _q < 4; ++_q)                                          \
            (BMB)[_q] = *(const v16h*)(bbase[_q] + (size_t)(KC) * 512);         \
    } while (0)

#define WMMA_CHUNK(AB, BMB)                                                     \
    do {                                                                        \
        _Pragma("unroll")                                                       \
        for (int _q = 0; _q < 4; ++_q)                                          \
            acc[_q] = __builtin_amdgcn_wmma_f32_16x16x32_f16(                   \
                false, (AB).v, false, (BMB)[_q], (short)0, acc[_q],             \
                false, false);                                                  \
    } while (0)

// ---------------------------------------------------------------------------
// Input projection GEMM: xp[dir][r=s*B+b][4096] = X16[r,:] @ W_ih^T + bias
// grid (16, 2048, 2), block 128 (4 waves); wave = 1 M-tile x 4 N-tiles.
// Register double-buffered K-loop; sched_barriers pin prefetch ahead of WMMA.
// ---------------------------------------------------------------------------
__global__ void __launch_bounds__(128, 1)
igemm_kernel(const _Float16* __restrict__ X16,
             const _Float16* __restrict__ WihPf,
             const _Float16* __restrict__ WihPb,
             const float* __restrict__ b_fw,
             const float* __restrict__ b_bw,
             float* __restrict__ xpf,
             float* __restrict__ xpb) {
    int dir = blockIdx.z;
    const _Float16* Wp = dir ? WihPb : WihPf;
    const float* bias  = dir ? b_bw : b_fw;
    float* xp          = dir ? xpb : xpf;

    int wave = threadIdx.x >> 5;
    int lane = threadIdx.x & 31;
    int ng = blockIdx.x * 4 + wave;   // 0..63 (group of 4 N-tiles)
    int mt = blockIdx.y;              // 0..2047

    int mrow = mt * 16 + (lane & 15);
    const _Float16* arow = X16 + (size_t)mrow * II;
    int aoff = (lane < 16) ? 0 : 8;

    const _Float16* bbase[4];
#pragma unroll
    for (int q = 0; q < 4; ++q)
        bbase[q] = Wp + ((size_t)(ng * 4 + q) * 32) * 512 + (size_t)lane * 16;

    v8f acc[4] = {v8f{}, v8f{}, v8f{}, v8f{}};

    V16U a0, a1;
    v16h bm0[4], bm1[4];
    LOAD_CHUNK(a0, bm0, 0);
    for (int kc = 0; kc < 30; kc += 2) {
        LOAD_CHUNK(a1, bm1, kc + 1);
        __builtin_amdgcn_sched_barrier(0);
        WMMA_CHUNK(a0, bm0);
        LOAD_CHUNK(a0, bm0, kc + 2);
        __builtin_amdgcn_sched_barrier(0);
        WMMA_CHUNK(a1, bm1);
    }
    LOAD_CHUNK(a1, bm1, 31);
    __builtin_amdgcn_sched_barrier(0);
    WMMA_CHUNK(a0, bm0);
    WMMA_CHUNK(a1, bm1);

#pragma unroll
    for (int q = 0; q < 4; ++q) {
        int n = (ng * 4 + q) * 16 + (lane & 15);
        float bb = bias[n];
#pragma unroll
        for (int j = 0; j < 8; ++j) {
            int m = mt * 16 + ((lane < 16) ? j : j + 8);
            xp[(size_t)m * G4 + n] = acc[q][j] + bb;
        }
    }
}

// ---------------------------------------------------------------------------
// Persistent recurrent kernel: loops over all 512 timesteps internally with a
// device-scope grid barrier between steps (64 co-resident blocks).
// grid (8, 4, 2), block 256 (8 waves). Wave owns (16 batches x 16 hidden cols),
// accumulates 4 gate tiles (n, n+H, n+2H, n+3H). Cell state c lives in regs.
// h double-buffered in global f16. xp gate rows are PREFETCHED (no registers)
// at step start via global_prefetch, loaded after the K-loop (L2/WGP$ hit).
// ---------------------------------------------------------------------------
__global__ void __launch_bounds__(256, 1)
lstm_persist_kernel(const _Float16* __restrict__ WhhPf,
                    const _Float16* __restrict__ WhhPb,
                    const float* __restrict__ xpf,
                    const float* __restrict__ xpb,
                    _Float16* __restrict__ h16,
                    const float* __restrict__ mask,
                    float* __restrict__ out,
                    int* __restrict__ bar) {
    int dir = blockIdx.z;
    const _Float16* Wp = dir ? WhhPb : WhhPf;
    const float* xp    = dir ? xpb : xpf;

    int wave = threadIdx.x >> 5;
    int lane = threadIdx.x & 31;
    int jn = blockIdx.x * 8 + wave;   // hidden n-tile 0..63
    int mt = blockIdx.y;              // batch tile 0..3

    int mrow = mt * 16 + (lane & 15);
    int aoff = (lane < 16) ? 0 : 8;
    int n = jn * 16 + (lane & 15);
    float mk = dir ? mask[n] : 1.0f;

    const _Float16* bbase[4];
#pragma unroll
    for (int q = 0; q < 4; ++q)
        bbase[q] = Wp + ((size_t)(q * 64 + jn) * 32) * 512 + (size_t)lane * 16;

    // per-lane (row, gate) mapping for the register-free xp prefetch:
    // lane l covers rows mt*16+(l&7) and mt*16+8+(l&7), gate l>>3
    int pf_row  = mt * 16 + (lane & 7);
    int pf_gate = lane >> 3;

    // cell state held entirely in registers for all 512 steps
    float creg[8];
#pragma unroll
    for (int j = 0; j < 8; ++j) creg[j] = 0.0f;

    const int nblk = 8 * 4 * 2;   // gridDim product = 64

    for (int t = 0; t < SS; ++t) {
        int s = dir ? (SS - 1 - t) : t;
        const _Float16* hin  = h16 + ((size_t)dir * 2 + (t & 1))       * (BB * HH);
        _Float16*       hout = h16 + ((size_t)dir * 2 + ((t & 1) ^ 1)) * (BB * HH);
        const _Float16* arow = hin + (size_t)mrow * HH;

        // register-free prefetch of this step's xp gate segments: each lane
        // prefetches one 64B segment; two instructions cover all 16 rows x 4
        // gates. Fetch latency hides under the K-loop GEMM.
        {
            const float* pa = xp + ((size_t)s * BB + pf_row) * G4
                                 + (size_t)pf_gate * HH + (size_t)jn * 16;
            __builtin_prefetch(pa, 0, 3);
            __builtin_prefetch(pa + (size_t)8 * G4, 0, 3);
        }

        v8f acc[4] = {v8f{}, v8f{}, v8f{}, v8f{}};
        V16U a0, a1;
        v16h bm0[4], bm1[4];
        LOAD_CHUNK(a0, bm0, 0);
        for (int kc = 0; kc < 30; kc += 2) {
            LOAD_CHUNK(a1, bm1, kc + 1);
            __builtin_amdgcn_sched_barrier(0);
            WMMA_CHUNK(a0, bm0);
            LOAD_CHUNK(a0, bm0, kc + 2);
            __builtin_amdgcn_sched_barrier(0);
            WMMA_CHUNK(a1, bm1);
        }
        LOAD_CHUNK(a1, bm1, 31);
        __builtin_amdgcn_sched_barrier(0);
        WMMA_CHUNK(a0, bm0);
        WMMA_CHUNK(a1, bm1);

        // gate pre-activations (prefetched above -> near-cache hits)
        float xg0[8], xg1[8], xg2[8], xg3[8];
#pragma unroll
        for (int j = 0; j < 8; ++j) {
            int m = mt * 16 + ((lane < 16) ? j : j + 8);
            const float* xr = xp + ((size_t)s * BB + m) * G4;
            xg0[j] = xr[n];
            xg1[j] = xr[HH + n];
            xg2[j] = xr[2 * HH + n];
            xg3[j] = xr[3 * HH + n];
        }

        // cell update + outputs
#pragma unroll
        for (int j = 0; j < 8; ++j) {
            int m = mt * 16 + ((lane < 16) ? j : j + 8);
            float ig = sigmf(acc[0][j] + xg0[j]);
            float fg = sigmf(acc[1][j] + xg1[j]);
            float gg = tanhfast(acc[2][j] + xg2[j]);
            float og = sigmf(acc[3][j] + xg3[j]);
            float cn = fg * creg[j] + ig * gg;
            creg[j] = cn;
            float h = og * tanhfast(cn);
            out[((size_t)m * SS + s) * (2 * HH) + (size_t)dir * HH + n] = h;
            hout[(size_t)m * HH + n] = (_Float16)(h * mk);
        }

        // device-scope grid barrier: release stores, arrive, spin, acquire
        __threadfence();
        __syncthreads();
        if (threadIdx.x == 0) {
            __hip_atomic_fetch_add(bar, 1, __ATOMIC_RELEASE, __HIP_MEMORY_SCOPE_AGENT);
            int target = nblk * (t + 1);
            while (__hip_atomic_load(bar, __ATOMIC_ACQUIRE, __HIP_MEMORY_SCOPE_AGENT) < target) {
                __builtin_amdgcn_s_sleep(2);
            }
        }
        __syncthreads();
        __threadfence();
    }
}

// ---------------------------------------------------------------------------
extern "C" void kernel_launch(void* const* d_in, const int* in_sizes, int n_in,
                              void* d_out, int out_size, void* d_ws, size_t ws_size,
                              hipStream_t stream) {
    (void)in_sizes; (void)n_in; (void)out_size; (void)ws_size;

    const float* x      = (const float*)d_in[0];
    const float* Wih_fw = (const float*)d_in[1];
    const float* Whh_fw = (const float*)d_in[2];
    const float* b_fw   = (const float*)d_in[3];
    const float* Wih_bw = (const float*)d_in[4];
    const float* Whh_bw = (const float*)d_in[5];
    const float* b_bw   = (const float*)d_in[6];
    const float* mask   = (const float*)d_in[7];
    float* out = (float*)d_out;

    // Workspace carve-out (256B aligned chunks)
    char* base = (char*)d_ws;
    size_t off = 0;
    auto carve = [&](size_t bytes) -> void* {
        void* r = base + off;
        off = (off + bytes + 255) & ~(size_t)255;
        return r;
    };
    _Float16* X16   = (_Float16*)carve((size_t)SS * BB * II * 2);   // 64 MiB
    _Float16* WihPf = (_Float16*)carve((size_t)G4 * HH * 2);        // 8 MiB
    _Float16* WihPb = (_Float16*)carve((size_t)G4 * HH * 2);
    _Float16* WhhPf = (_Float16*)carve((size_t)G4 * HH * 2);
    _Float16* WhhPb = (_Float16*)carve((size_t)G4 * HH * 2);
    float*    xpf   = (float*)carve((size_t)SS * BB * G4 * 4);      // 512 MiB
    float*    xpb   = (float*)carve((size_t)SS * BB * G4 * 4);      // 512 MiB
    _Float16* h16   = (_Float16*)carve((size_t)2 * 2 * BB * HH * 2);
    int*      bar   = (int*)carve(256);

    // Phase 0: convert x, pack weights, init state + barrier counter
    {
        size_t total = (size_t)SS * BB * II;
        conv_x_kernel<<<(unsigned)((total + 255) / 256), 256, 0, stream>>>(x, X16);
    }
    {
        int total = G4 * HH;
        int blks = (total + 255) / 256;
        pack_w_kernel<<<blks, 256, 0, stream>>>(Wih_fw, WihPf);
        pack_w_kernel<<<blks, 256, 0, stream>>>(Wih_bw, WihPb);
        pack_w_kernel<<<blks, 256, 0, stream>>>(Whh_fw, WhhPf);
        pack_w_kernel<<<blks, 256, 0, stream>>>(Whh_bw, WhhPb);
    }
    init_state_kernel<<<(2 * 2 * BB * HH + 255) / 256, 256, 0, stream>>>(h16, bar);

    // Phase 1: input projection GEMM for both directions (WMMA)
    igemm_kernel<<<dim3(16, (SS * BB) / 16, 2), 128, 0, stream>>>(
        X16, WihPf, WihPb, b_fw, b_bw, xpf, xpb);

    // Phase 2: single persistent launch runs the entire 512-step recurrence
    lstm_persist_kernel<<<dim3(8, 4, 2), 256, 0, stream>>>(
        WhhPf, WhhPb, xpf, xpb, h16, mask, out, bar);
}
